// BitLinear_38190849196566
// MI455X (gfx1250) — 8-bit matrix operand order repaired and checked by reading, compiled, not run on hardware
//
#include <hip/hip_runtime.h>

typedef __attribute__((ext_vector_type(8))) int v8i;

constexpr int MTOT = 8 * 4096;
constexpr int KDIM = 1024;
constexpr int NDIM = 1024;

constexpr size_t WS_SUMABS = 0;
constexpr size_t WS_ALPHA  = 256;
constexpr size_t WS_SCALEX = 4608;
constexpr size_t WS_WQ     = 135680;
constexpr size_t WS_XQ     = 1184256;

__global__ void init_ws_kernel(float* sumAbs) {
    if (threadIdx.x == 0) *sumAbs = 0.0f;
}

__global__ __launch_bounds__(256) void wabs_sum_kernel(const float* __restrict__ w,
                                                       float* __restrict__ sumAbs) {
    __shared__ float red[256];
    const int t = threadIdx.x;
    const float4 v = reinterpret_cast<const float4*>(w)[(size_t)blockIdx.x * 256 + t];
    red[t] = fabsf(v.x) + fabsf(v.y) + fabsf(v.z) + fabsf(v.w);
    __syncthreads();
    for (int off = 128; off > 0; off >>= 1) {
        if (t < off) red[t] += red[t + off];
        __syncthreads();
    }
    if (t == 0) atomicAdd(sumAbs, red[0]);
}

__global__ __launch_bounds__(256) void wquant_kernel(const float* __restrict__ w,
                                                     const float* __restrict__ sumAbs,
                                                     signed char* __restrict__ wq,
                                                     float* __restrict__ alpha) {
    __shared__ float red[256];
    const int t = threadIdx.x;
    const int o = blockIdx.x;
    const float4 v = reinterpret_cast<const float4*>(w + (size_t)o * KDIM)[t];
    red[t] = fabsf(v.x) + fabsf(v.y) + fabsf(v.z) + fabsf(v.w);
    __syncthreads();
    for (int off = 128; off > 0; off >>= 1) {
        if (t < off) red[t] += red[t + off];
        __syncthreads();
    }
    const float a = red[0] * (1.0f / (float)KDIM);
    const float thresh = 0.05f * (*sumAbs) * (1.0f / ((float)KDIM * (float)NDIM));
    int q0 = (fabsf(v.x) < thresh) ? 0 : ((v.x > 0.0f) ? 1 : -1);
    int q1 = (fabsf(v.y) < thresh) ? 0 : ((v.y > 0.0f) ? 1 : -1);
    int q2 = (fabsf(v.z) < thresh) ? 0 : ((v.z > 0.0f) ? 1 : -1);
    int q3 = (fabsf(v.w) < thresh) ? 0 : ((v.w > 0.0f) ? 1 : -1);
    const int packed = (q0 & 255) | ((q1 & 255) << 8) | ((q2 & 255) << 16) | ((q3 & 255) << 24);
    reinterpret_cast<int*>(wq + (size_t)o * KDIM)[t] = packed;
    if (t == 0) alpha[o] = a;
}

__global__ __launch_bounds__(256) void xquant_kernel(const float* __restrict__ x,
                                                     signed char* __restrict__ xq,
                                                     float* __restrict__ scaleX) {
    __shared__ float red[256];
    const int t = threadIdx.x;
    const size_t row = blockIdx.x;
    const float4 v = reinterpret_cast<const float4*>(x + row * KDIM)[t];
    red[t] = fmaxf(fmaxf(fabsf(v.x), fabsf(v.y)), fmaxf(fabsf(v.z), fabsf(v.w)));
    __syncthreads();
    for (int off = 128; off > 0; off >>= 1) {
        if (t < off) red[t] = fmaxf(red[t], red[t + off]);
        __syncthreads();
    }
    const float ma = fmaxf(red[0], 1e-6f);
    const float inv = 7.0f / ma;
    int q0 = max(-8, min(7, (int)rintf(v.x * inv)));
    int q1 = max(-8, min(7, (int)rintf(v.y * inv)));
    int q2 = max(-8, min(7, (int)rintf(v.z * inv)));
    int q3 = max(-8, min(7, (int)rintf(v.w * inv)));
    const int packed = (q0 & 255) | ((q1 & 255) << 8) | ((q2 & 255) << 16) | ((q3 & 255) << 24);
    reinterpret_cast<int*>(xq + row * KDIM)[t] = packed;
    if (t == 0) scaleX[row] = ma * (1.0f / 7.0f);
}

__global__ __launch_bounds__(256) void gemm_iu8_kernel(const signed char* __restrict__ xq,
                                                       const signed char* __restrict__ wq,
                                                       const float* __restrict__ scaleX,
                                                       const float* __restrict__ alpha,
                                                       const float* __restrict__ bias,
                                                       float* __restrict__ out) {
    __shared__ int lb[512];

    const int t    = threadIdx.x;
    const int lane = t & 31;
    const int n    = lane & 15;
    const int hi   = lane >> 4;
    const int wave = t >> 5;

    const int nbase = blockIdx.x << 5;
    const int waveM = (blockIdx.y << 9) + (wave << 6);

    unsigned aoff[4];
#pragma unroll
    for (int mt = 0; mt < 4; ++mt)
        aoff[mt] = (unsigned)(waveM + (mt << 4) + n) * (unsigned)KDIM + (unsigned)(hi << 4);

    const unsigned boff = (unsigned)(nbase + (t >> 3)) * (unsigned)KDIM +
                          (unsigned)((t & 7) << 3);
    const unsigned ldsdst =
        (unsigned)(unsigned long long)(&lb[t << 1]);

    v8i acc[4][2];
#pragma unroll
    for (int i = 0; i < 4; ++i) {
        acc[i][0] = v8i{0, 0, 0, 0, 0, 0, 0, 0};
        acc[i][1] = v8i{0, 0, 0, 0, 0, 0, 0, 0};
    }

#pragma unroll 1
    for (int kt = 0; kt < KDIM; kt += 64) {
        {
            const signed char* bsrc = wq + kt + boff;
            asm volatile("global_load_async_to_lds_b64 %0, %1, off"
                         :
                         : "v"(ldsdst), "v"(bsrc)
                         : "memory");
            asm volatile("s_wait_asynccnt 0x0" ::: "memory");
        }
        __syncthreads();

        const int* bp0 = &lb[(n << 4) + (hi << 2)];
        const int* bp1 = &lb[((16 + n) << 4) + (hi << 2)];
        const int4 a0 = *reinterpret_cast<const int4*>(bp0);
        const int4 a1 = *reinterpret_cast<const int4*>(bp0 + 8);
        const int4 a2 = *reinterpret_cast<const int4*>(bp1);
        const int4 a3 = *reinterpret_cast<const int4*>(bp1 + 8);
        const v8i bf0 = v8i{a0.x, a0.y, a0.z, a0.w, a1.x, a1.y, a1.z, a1.w};
        const v8i bf1 = v8i{a2.x, a2.y, a2.z, a2.w, a3.x, a3.y, a3.z, a3.w};
        __syncthreads();

        const int pf = (kt + 64 < KDIM) ? 64 : 0;

#pragma unroll
        for (int mt = 0; mt < 4; ++mt) {
            const signed char* ap = xq + kt + aoff[mt];
            const int2 c0 = *reinterpret_cast<const int2*>(ap);
            const int2 c1 = *reinterpret_cast<const int2*>(ap + 8);
            const int2 c2 = *reinterpret_cast<const int2*>(ap + 32);
            const int2 c3 = *reinterpret_cast<const int2*>(ap + 40);
            __builtin_prefetch(ap + pf, 0, 3);
            const v8i af = v8i{c0.x, c0.y, c1.x, c1.y, c2.x, c2.y, c3.x, c3.y};
            acc[mt][0] = __builtin_amdgcn_wmma_i32_16x16x64_iu8(
                true, af, true, bf0, acc[mt][0], false, false);
            acc[mt][1] = __builtin_amdgcn_wmma_i32_16x16x64_iu8(
                true, af, true, bf1, acc[mt][1], false, false);
        }
    }

    const int col0 = nbase + n;
    const int col1 = nbase + 16 + n;
    const float aC0 = alpha[col0];
    const float bC0 = bias[col0];
    const float aC1 = alpha[col1];
    const float bC1 = bias[col1];
#pragma unroll
    for (int mt = 0; mt < 4; ++mt) {
#pragma unroll
        for (int r = 0; r < 8; ++r) {
            const int row = waveM + (mt << 4) + (hi << 3) + r;
            const float sX = scaleX[row];
            out[(size_t)row * NDIM + col0] = (float)acc[mt][0][r] * sX * aC0 + bC0;
            out[(size_t)row * NDIM + col1] = (float)acc[mt][1][r] * sX * aC1 + bC1;
        }
    }
}

extern "C" void kernel_launch(void* const* d_in, const int* in_sizes, int n_in,
                              void* d_out, int out_size, void* d_ws, size_t ws_size,
                              hipStream_t stream) {
    const float* x    = (const float*)d_in[0];
    const float* w    = (const float*)d_in[1];
    const float* bias = (const float*)d_in[2];
    float* out = (float*)d_out;

    char* ws = (char*)d_ws;
    float* sumAbs       = (float*)(ws + WS_SUMABS);
    float* alpha        = (float*)(ws + WS_ALPHA);
    float* scaleX       = (float*)(ws + WS_SCALEX);
    signed char* wq     = (signed char*)(ws + WS_WQ);
    signed char* xq     = (signed char*)(ws + WS_XQ);

    init_ws_kernel<<<1, 1, 0, stream>>>(sumAbs);
    wabs_sum_kernel<<<(NDIM * KDIM) / 1024, 256, 0, stream>>>(w, sumAbs);
    wquant_kernel<<<NDIM, 256, 0, stream>>>(w, sumAbs, wq, alpha);
    xquant_kernel<<<MTOT, 256, 0, stream>>>(x, xq, scaleX);

    dim3 grid(NDIM / 32, MTOT / 512);
    gemm_iu8_kernel<<<grid, 256, 0, stream>>>(xq, wq, scaleX, alpha, bias, out);
}
